// RefinedAMSAM_17239998726609
// MI455X (gfx1250) — matrix operand order repaired and checked by reading, not compiled, not run on hardware
//
#include <hip/hip_runtime.h>
#include <hip/hip_bf16.h>

// ---------------- problem constants ----------------
constexpr int NS   = 3;      // scales {1,2,4}
constexpr int NB   = 4;      // batch
constexpr int CI   = 64;     // CIN
constexpr int CO   = 128;    // COUT
constexpr int HW   = 56;     // H = W
constexpr int NPIX = HW * HW;       // 3136
constexpr int NQKV = 160;           // 16 Q + 16 K + 128 V rows
constexpr int KB   = NPIX / 32;     // 98 key blocks
constexpr float BN_EPS = 1e-5f;

typedef __attribute__((ext_vector_type(16))) _Float16 v16h;
typedef __attribute__((ext_vector_type(8)))  _Float16 v8h;
typedef __attribute__((ext_vector_type(8)))  float    v8f;
typedef __attribute__((ext_vector_type(4)))  int      v4i;
typedef __attribute__((ext_vector_type(8)))  int      v8i;

// gfx1250 async global->LDS path (ASYNCcnt), if this toolchain exposes it.
#if defined(__has_builtin)
#if __has_builtin(__builtin_amdgcn_global_load_async_to_lds_b128) && \
    __has_builtin(__builtin_amdgcn_s_wait_asynccnt)
#define USE_ASYNC_LDS 1
#endif
#endif
#ifndef USE_ASYNC_LDS
#define USE_ASYNC_LDS 0
#endif
#define AS1 __attribute__((address_space(1)))
#define AS3 __attribute__((address_space(3)))

static __device__ inline v16h cat8(v8h lo, v8h hi) {
    return __builtin_shufflevector(lo, hi, 0,1,2,3,4,5,6,7,8,9,10,11,12,13,14,15);
}
// lane <-> lane^16 exchange via ds_swizzle (group-of-32: and=0x1f, xor=0x10)
static __device__ inline float swz16f(float x) {
    return __int_as_float(__builtin_amdgcn_ds_swizzle(__float_as_int(x), 0x401F));
}
// A-operand (16x32 f16): lane row = lane%16, K = 16*hsel..+15 (contiguous)
static __device__ inline v16h loadA(const _Float16* row, int hsel) {
    return cat8(*(const v8h*)(row + hsel * 16), *(const v8h*)(row + hsel * 16 + 8));
}

// ============================================================
// Kernel 0: pack weights to f16 / fold BN.
// ============================================================
__global__ __launch_bounds__(256) void pack_kernel(
    const float* __restrict__ pw_w,
    const float* __restrict__ bn_g, const float* __restrict__ bn_b,
    const float* __restrict__ bn_m, const float* __restrict__ bn_v,
    const float* __restrict__ wq_w, const float* __restrict__ wq_b,
    const float* __restrict__ wk_w, const float* __restrict__ wk_b,
    const float* __restrict__ wv_w, const float* __restrict__ wv_b,
    const float* __restrict__ proj_w,
    _Float16* __restrict__ Wpw, float* __restrict__ bnA, float* __restrict__ bnB,
    _Float16* __restrict__ Wqkv, float* __restrict__ Bqkv, _Float16* __restrict__ Wproj)
{
    const int id = blockIdx.x * 256 + threadIdx.x;
    if (id < NS * CO * CI) Wpw[id] = (_Float16)pw_w[id];
    if (id < CO * CI)      Wproj[id] = (_Float16)proj_w[id];
    if (id < NS * CO) {
        float a = bn_g[id] * rsqrtf(bn_v[id] + BN_EPS);
        bnA[id] = a;
        bnB[id] = bn_b[id] - bn_m[id] * a;
    }
    if (id < NS * NQKV) {
        int s = id / NQKV, o = id % NQKV;
        Bqkv[id] = (o < 16) ? wq_b[s * 16 + o]
                 : (o < 32) ? wk_b[s * 16 + o - 16]
                            : wv_b[s * CO + o - 32];
    }
    if (id < NS * NQKV * CO) {
        int s = id / (NQKV * CO), r = id % (NQKV * CO), o = r / CO, c = r % CO;
        float w = (o < 16) ? wq_w[(size_t)(s * 16 + o) * CO + c]
                : (o < 32) ? wk_w[(size_t)(s * 16 + o - 16) * CO + c]
                           : wv_w[((size_t)s * CO + o - 32) * CO + c];
        Wqkv[id] = (_Float16)w;
    }
}

// fold pw bias through BN shift: bnB += pw_b * bnA
__global__ __launch_bounds__(128) void fixb_kernel(
    const float* __restrict__ pwb, const float* __restrict__ A, float* __restrict__ Bv)
{
    int i = blockIdx.x * 128 + threadIdx.x;
    if (i < NS * CO) Bv[i] += pwb[i] * A[i];
}

// ============================================================
// Kernel 1: depthwise (VALU) + pointwise/BN/ReLU (WMMA)
//   -> Ff[sb][n][128] f16
// grid (25, NB, NS), 256 threads, 128-px tile
// ============================================================
__global__ __launch_bounds__(256) void feat_kernel(
    const float* __restrict__ x,
    const float* __restrict__ dw_w, const float* __restrict__ dw_b,
    const _Float16* __restrict__ Wpw,
    const float* __restrict__ bnA, const float* __restrict__ bnB,
    _Float16* __restrict__ Ff)
{
    __shared__ __align__(64) _Float16 Wl[CO * CI];    // [o][ci]   16KB
    __shared__ __align__(64) _Float16 dwB[128 * CI];  // [px][ci]  16KB
    const int s = blockIdx.z, b = blockIdx.y, sb = s * NB + b;
    const int n0 = blockIdx.x * 128;
    const int t = threadIdx.x;
    const int dil = 1 << s;                          // 1,2,4
    const float* xb = x + (size_t)b * CI * NPIX;

    {   // stage pw weights (contiguous copy)
        const int4* src = (const int4*)(Wpw + (size_t)s * CO * CI);
        #pragma unroll
        for (int i = 0; i < 4; ++i) ((int4*)Wl)[t + 256 * i] = src[t + 256 * i];
    }
    // depthwise 3x3 dilated -> f16 B-layout
    for (int r = 0; r < 32; ++r) {
        int idx = r * 256 + t;
        int ci = idx >> 7, px = idx & 127, n = n0 + px;
        float acc = 0.f;
        if (n < NPIX) {
            int h = n / HW, w = n % HW;
            acc = dw_b[s * CI + ci];
            const float* wr = dw_w + (size_t)(s * CI + ci) * 9;
            const float* xc = xb + (size_t)ci * NPIX;
            #pragma unroll
            for (int ky = 0; ky < 3; ++ky) {
                int hh = h + (ky - 1) * dil;
                if ((unsigned)hh < (unsigned)HW) {
                    #pragma unroll
                    for (int kx = 0; kx < 3; ++kx) {
                        int ww = w + (kx - 1) * dil;
                        if ((unsigned)ww < (unsigned)HW)
                            acc += wr[ky * 3 + kx] * xc[hh * HW + ww];
                    }
                }
            }
        }
        dwB[px * CI + ci] = (_Float16)acc;
    }
    __syncthreads();

    const int wave = t >> 5, lane = t & 31, col = lane & 15, hsel = lane >> 4;
    for (int tile = wave; tile < 64; tile += 8) {     // 8 o-tiles x 8 px-tiles
        int o0 = (tile >> 3) * 16, px0 = (tile & 7) * 16;
        v8f acc = (v8f){};
        #pragma unroll
        for (int ks = 0; ks < 2; ++ks) {
            v16h aw = loadA(&Wl[(o0 + col) * CI + ks * 32], hsel);
            v16h bf = *(const v16h*)(&dwB[(px0 + col) * CI + ks * 32 + hsel * 16]);
            acc = __builtin_amdgcn_wmma_f32_16x16x32_f16(false, aw, false, bf,
                                                         (short)0, acc, false, false);
        }
        int n = n0 + px0 + col;
        if (n < NPIX) {
            #pragma unroll
            for (int v = 0; v < 8; ++v) {
                int o = o0 + v + 8 * hsel;
                float f = fmaxf(acc[v] * bnA[s * CO + o] + bnB[s * CO + o], 0.f);
                Ff[((size_t)sb * NPIX + n) * CO + o] = (_Float16)f;
            }
        }
    }
}

// ============================================================
// Kernel 2a: partial channel sums.  grid (8, NS*NB), 128 thr
// ============================================================
__global__ __launch_bounds__(128) void avgp_kernel(
    const _Float16* __restrict__ Ff, float* __restrict__ part)
{
    const int chunk = blockIdx.x, sb = blockIdx.y, c = threadIdx.x;
    const int nst = chunk * (NPIX / 8);
    float s = 0.f;
    const _Float16* src = Ff + ((size_t)sb * NPIX + nst) * CO + c;
    for (int i = 0; i < NPIX / 8; ++i) s += (float)src[(size_t)i * CO];
    part[((size_t)sb * 8 + chunk) * CO + c] = s;
}

// ============================================================
// Kernel 2b: channel gate ca = sigmoid(fc2(relu(fc1(avg))))
// ============================================================
__global__ __launch_bounds__(128) void ca_kernel(
    const float* __restrict__ part,
    const float* __restrict__ fc1_w, const float* __restrict__ fc1_b,
    const float* __restrict__ fc2_w, const float* __restrict__ fc2_b,
    float* __restrict__ ca)
{
    const int sb = blockIdx.x, s = sb / NB, t = threadIdx.x;
    __shared__ float av[CO];
    __shared__ float hid[8];
    float sum = 0.f;
    #pragma unroll
    for (int k = 0; k < 8; ++k) sum += part[((size_t)sb * 8 + k) * CO + t];
    av[t] = sum / (float)NPIX;
    __syncthreads();
    if (t < 8) {
        float a = fc1_b[s * 8 + t];
        const float* w = fc1_w + (size_t)(s * 8 + t) * CO;
        for (int c = 0; c < CO; ++c) a += w[c] * av[c];
        hid[t] = fmaxf(a, 0.f);
    }
    __syncthreads();
    float a = fc2_b[s * CO + t];
    const float* w = fc2_w + (size_t)(s * CO + t) * 8;
    #pragma unroll
    for (int h = 0; h < 8; ++h) a += w[h] * hid[h];
    ca[(size_t)sb * CO + t] = 1.f / (1.f + __expf(-a));
}

// ============================================================
// Kernel 3: QKV projections via WMMA.
//   Qp/Kt[sb][n][32] (cols 16..31 zero), Vt[sb][kb][128][32]
// grid (49, NB, NS), 256 threads, 64-px tile
// ============================================================
__global__ __launch_bounds__(256) void qkv_kernel(
    const _Float16* __restrict__ Ff,
    const _Float16* __restrict__ Wqkv, const float* __restrict__ Bqkv,
    _Float16* __restrict__ Qp, _Float16* __restrict__ Kt, _Float16* __restrict__ Vt)
{
    __shared__ __align__(64) _Float16 Wl[NQKV * CO];  // 40KB
    __shared__ __align__(64) _Float16 Fl[64 * CO];    // 16KB
    const int s = blockIdx.z, b = blockIdx.y, sb = s * NB + b;
    const int n0 = blockIdx.x * 64;
    const int t = threadIdx.x;
    {
        const int4* src = (const int4*)(Wqkv + (size_t)s * NQKV * CO);
        #pragma unroll
        for (int i = 0; i < 10; ++i) ((int4*)Wl)[t + 256 * i] = src[t + 256 * i];
        const int4* fs = (const int4*)(Ff + ((size_t)sb * NPIX + n0) * CO);
        #pragma unroll
        for (int i = 0; i < 4; ++i) ((int4*)Fl)[t + 256 * i] = fs[t + 256 * i];
    }
    __syncthreads();

    const int wave = t >> 5, lane = t & 31, col = lane & 15, hsel = lane >> 4;
    _Float16* Qg = Qp + (size_t)sb * NPIX * 32;
    _Float16* Kg = Kt + (size_t)sb * NPIX * 32;
    _Float16* Vg = Vt + (size_t)sb * NPIX * CO;
    for (int tile = wave; tile < 40; tile += 8) {     // 10 o-tiles x 4 px-tiles
        int o0 = (tile >> 2) * 16, px0 = (tile & 3) * 16;
        v8f acc = (v8f){};
        #pragma unroll
        for (int ks = 0; ks < 4; ++ks) {
            v16h aw = loadA(&Wl[(o0 + col) * CO + ks * 32], hsel);
            v16h bf = *(const v16h*)(&Fl[(px0 + col) * CO + ks * 32 + hsel * 16]);
            acc = __builtin_amdgcn_wmma_f32_16x16x32_f16(false, aw, false, bf,
                                                         (short)0, acc, false, false);
        }
        int n = n0 + px0 + col;
        if (o0 == 0) {
            #pragma unroll
            for (int v = 0; v < 8; ++v) {
                int o = v + 8 * hsel;
                Qg[(size_t)n * 32 + o]      = (_Float16)(acc[v] + Bqkv[s * NQKV + o]);
                Qg[(size_t)n * 32 + o + 16] = (_Float16)0.f;
            }
        } else if (o0 == 16) {
            #pragma unroll
            for (int v = 0; v < 8; ++v) {
                int qd = v + 8 * hsel;
                Kg[(size_t)n * 32 + qd]      = (_Float16)(acc[v] + Bqkv[s * NQKV + 16 + qd]);
                Kg[(size_t)n * 32 + qd + 16] = (_Float16)0.f;
            }
        } else {
            #pragma unroll
            for (int v = 0; v < 8; ++v) {
                int o = o0 + v + 8 * hsel, c = o - 32;
                Vg[(((size_t)(n >> 5)) * CO + c) * 32 + (n & 31)] =
                    (_Float16)(acc[v] + Bqkv[s * NQKV + o]);
            }
        }
    }
}

// ============================================================
// Kernel 4: transposed flash attention, double-buffered staging.
// Async global->LDS (ASYNCcnt) when available, else sync dual-buffer.
// grid (25, NB, NS), 256 threads (8 waves, 1 query-tile each)
// ============================================================
__global__ __launch_bounds__(256) void attn_kernel(
    const _Float16* __restrict__ Qp,   // [sb][n][32]
    const _Float16* __restrict__ Kt,   // [sb][n][32]
    const _Float16* __restrict__ Vt,   // [sb][kb][128][32]
    const float*    __restrict__ ca,   // [sb][128]
    float* __restrict__ af2)           // [sb][n][128]
{
    __shared__ __align__(64) _Float16 Klds[2][32 * 32];   // 2 x 2KB
    __shared__ __align__(64) _Float16 Vlds[2][CO * 32];   // 2 x 8KB

    const int s = blockIdx.z, b = blockIdx.y, sb = s * NB + b;
    const int t = threadIdx.x;
    const int wave = t >> 5, lane = t & 31;
    const int col = lane & 15, hsel = lane >> 4;

    const int qtile = blockIdx.x * 8 + wave;           // 196 valid of 200
    const bool valid = qtile < (NPIX / 16);            // wave-uniform

    const _Float16* Qg = Qp + (size_t)sb * NPIX * 32;
    const _Float16* Kg = Kt + (size_t)sb * NPIX * 32;
    const _Float16* Vg = Vt + (size_t)sb * NPIX * CO;

    // loop-invariant B operand: Q[query=col][qd = 16*hsel .. +15]
    v16h bq = {};
    if (valid)
        bq = *(const v16h*)(Qg + (size_t)(qtile * 16 + col) * 32 + hsel * 16);

    v8f oc[8];
    #pragma unroll
    for (int g = 0; g < 8; ++g) oc[g] = (v8f){};
    float rmax = -1e30f, rsum = 0.f;
    const float inv_sqrt_c = 0.08838834764831845f;     // 1/sqrt(128)
    const v8f zero8 = (v8f){};

#if USE_ASYNC_LDS
    auto stage_async = [&](int kb, int buf) {
        AS1 v4i* ksrc = (AS1 v4i*)(Kg + (size_t)kb * 1024);
        AS3 v4i* kdst = (AS3 v4i*)(&Klds[buf][0]);
        if (t < 128)                                    // K tile: 2KB
            __builtin_amdgcn_global_load_async_to_lds_b128(ksrc + t, kdst + t, 0, 0);
        AS1 v4i* vsrc = (AS1 v4i*)(Vg + (size_t)kb * CO * 32);
        AS3 v4i* vdst = (AS3 v4i*)(&Vlds[buf][0]);
        __builtin_amdgcn_global_load_async_to_lds_b128(vsrc + t, vdst + t, 0, 0);
        __builtin_amdgcn_global_load_async_to_lds_b128(vsrc + t + 256, vdst + t + 256, 0, 0);
    };
    stage_async(0, 0);
    __builtin_amdgcn_s_wait_asynccnt(0);
    __syncthreads();
#else
    int2 kreg = ((const int2*)Kg)[t];
    int4 vreg0, vreg1;
    {
        const int4* vs = (const int4*)Vg;
        vreg0 = vs[t]; vreg1 = vs[t + 256];
    }
    ((int2*)Klds[0])[t] = kreg;
    ((int4*)Vlds[0])[t] = vreg0;
    ((int4*)Vlds[0])[t + 256] = vreg1;
    __syncthreads();
#endif

    for (int kb = 0; kb < KB; ++kb) {
        const int cur = kb & 1;
        const bool more = (kb + 1 < KB);
#if USE_ASYNC_LDS
        if (more) stage_async(kb + 1, cur ^ 1);        // overlaps compute below
#else
        if (more) {                                     // early issue -> regs
            kreg = ((const int2*)(Kg + (size_t)(kb + 1) * 1024))[t];
            const int4* vs = (const int4*)(Vg + (size_t)(kb + 1) * CO * 32);
            vreg0 = vs[t]; vreg1 = vs[t + 256];
        }
        if (kb + 2 < KB) {                              // gfx1250 global_prefetch_b8
            __builtin_prefetch(Kg + (size_t)(kb + 2) * 1024 + t * 4, 0, 1);
            __builtin_prefetch(Vg + (size_t)(kb + 2) * CO * 32 + t * 16, 0, 1);
        }
#endif
        if (valid) {                                   // EXEC all-1 inside
            v16h ak0 = loadA(&Klds[cur][col * 32], hsel);          // keys 0..15
            v16h ak1 = loadA(&Klds[cur][(col + 16) * 32], hsel);   // keys 16..31
            v8f d0 = __builtin_amdgcn_wmma_f32_16x16x32_f16(false, ak0, false, bq,
                                                            (short)0, zero8, false, false);
            v8f d1 = __builtin_amdgcn_wmma_f32_16x16x32_f16(false, ak1, false, bq,
                                                            (short)0, zero8, false, false);
            // lane holds keys v+8*hsel (d0) and 16+v+8*hsel (d1) for query 'col'
            float tm = -1e30f;
            #pragma unroll
            for (int v = 0; v < 8; ++v) {
                d0[v] *= inv_sqrt_c; d1[v] *= inv_sqrt_c;
                tm = fmaxf(tm, fmaxf(d0[v], d1[v]));
            }
            tm = fmaxf(tm, swz16f(tm));                // combine with lane^16
            float nm  = fmaxf(rmax, tm);
            float scl = __expf(rmax - nm);
            rmax = nm;
            float p0[8], p1[8], ps = 0.f;
            #pragma unroll
            for (int v = 0; v < 8; ++v) {
                p0[v] = __expf(d0[v] - nm);
                p1[v] = __expf(d1[v] - nm);
                ps += p0[v] + p1[v];
            }
            ps += swz16f(ps);
            rsum = rsum * scl + ps;
            #pragma unroll
            for (int g = 0; g < 8; ++g) {
                #pragma unroll
                for (int v = 0; v < 8; ++v) oc[g][v] *= scl;
            }
            // B operand P^T: own 8 probs + partner(lane^16) 8 probs
            v8h ph0, ph1;
            #pragma unroll
            for (int v = 0; v < 8; ++v) { ph0[v] = (_Float16)p0[v]; ph1[v] = (_Float16)p1[v]; }
            v4i a0 = __builtin_bit_cast(v4i, ph0), a1 = __builtin_bit_cast(v4i, ph1);
            v4i e0, e1;
            #pragma unroll
            for (int i = 0; i < 4; ++i) {
                e0[i] = __builtin_amdgcn_ds_swizzle(a0[i], 0x401F);
                e1[i] = __builtin_amdgcn_ds_swizzle(a1[i], 0x401F);
            }
            v4i lo, hi;
            if (hsel) { lo = e1; hi = a1; }            // keys 16..23 | 24..31
            else      { lo = a0; hi = e0; }            // keys  0..7  |  8..15
            v16h bp = __builtin_bit_cast(v16h, __builtin_shufflevector(lo, hi, 0,1,2,3,4,5,6,7));
            // O^T += V-tile x P^T over 8 channel groups
            #pragma unroll
            for (int g = 0; g < 8; ++g) {
                v16h av = loadA(&Vlds[cur][(g * 16 + col) * 32], hsel);
                oc[g] = __builtin_amdgcn_wmma_f32_16x16x32_f16(false, av, false, bp,
                                                               (short)0, oc[g], false, false);
            }
        }
#if USE_ASYNC_LDS
        __builtin_amdgcn_s_wait_asynccnt(0);
        __syncthreads();
#else
        if (more) {
            ((int2*)Klds[cur ^ 1])[t] = kreg;
            ((int4*)Vlds[cur ^ 1])[t] = vreg0;
            ((int4*)Vlds[cur ^ 1])[t + 256] = vreg1;
        }
        __syncthreads();
#endif
    }

    if (valid) {
        float rinv = 1.f / rsum;
        int q = qtile * 16 + col;
        const float* cag = ca + (size_t)sb * CO;
        float* orow = af2 + ((size_t)sb * NPIX + q) * CO;
        #pragma unroll
        for (int g = 0; g < 8; ++g) {                  // ch = g*16 + 8*hsel + v
            v8f cav = *(const v8f*)(cag + g * 16 + hsel * 8);
            v8f r;
            #pragma unroll
            for (int v = 0; v < 8; ++v) {
                float af = oc[g][v] * rinv * cav[v];
                r[v] = af * af;
            }
            *(v8f*)(orow + g * 16 + hsel * 8) = r;     // 32B store
        }
    }
}

// ============================================================
// Kernel 5: out = proj(x) + sum_s af2[s]   (WMMA projection)
// grid (25, NB), 256 threads, 128-px tile
// ============================================================
__global__ __launch_bounds__(256) void fuse_kernel(
    const float* __restrict__ x,
    const _Float16* __restrict__ Wproj, const float* __restrict__ proj_b,
    const float* __restrict__ af2, float* __restrict__ out)
{
    __shared__ __align__(64) _Float16 Wl[CO * CI];    // 16KB
    __shared__ __align__(64) _Float16 xB[128 * CI];   // 16KB
    const int b = blockIdx.y, n0 = blockIdx.x * 128, t = threadIdx.x;
    {
        #pragma unroll
        for (int i = 0; i < 4; ++i)
            ((int4*)Wl)[t + 256 * i] = ((const int4*)Wproj)[t + 256 * i];
    }
    for (int r = 0; r < 32; ++r) {
        int idx = r * 256 + t;
        int ci = idx >> 7, px = idx & 127, n = n0 + px;
        float xv = (n < NPIX) ? x[((size_t)b * CI + ci) * NPIX + n] : 0.f;
        xB[px * CI + ci] = (_Float16)xv;
    }
    __syncthreads();

    const int wave = t >> 5, lane = t & 31, col = lane & 15, hsel = lane >> 4;
    const size_t sstride = (size_t)NB * NPIX * CO;
    for (int tile = wave; tile < 64; tile += 8) {
        int o0 = (tile >> 3) * 16, px0 = (tile & 7) * 16;
        v8f acc = (v8f){};
        #pragma unroll
        for (int ks = 0; ks < 2; ++ks) {
            v16h aw = loadA(&Wl[(o0 + col) * CI + ks * 32], hsel);
            v16h bf = *(const v16h*)(&xB[(px0 + col) * CI + ks * 32 + hsel * 16]);
            acc = __builtin_amdgcn_wmma_f32_16x16x32_f16(false, aw, false, bf,
                                                         (short)0, acc, false, false);
        }
        int n = n0 + px0 + col;
        if (n < NPIX) {
            #pragma unroll
            for (int v = 0; v < 8; ++v) {
                int o = o0 + v + 8 * hsel;
                float r = acc[v] + proj_b[o];
                size_t ai = ((size_t)b * NPIX + n) * CO + o;   // s = 0 index
                r += af2[ai] + af2[ai + sstride] + af2[ai + 2 * sstride];
                out[((size_t)b * CO + o) * NPIX + n] = r;
            }
        }
    }
}

// ============================================================
extern "C" void kernel_launch(void* const* d_in, const int* in_sizes, int n_in,
                              void* d_out, int out_size, void* d_ws, size_t ws_size,
                              hipStream_t stream)
{
    const float* x      = (const float*)d_in[0];
    const float* dw_w   = (const float*)d_in[1];
    const float* dw_b   = (const float*)d_in[2];
    const float* pw_w   = (const float*)d_in[3];
    const float* pw_b   = (const float*)d_in[4];
    const float* bn_g   = (const float*)d_in[5];
    const float* bn_b   = (const float*)d_in[6];
    const float* bn_m   = (const float*)d_in[7];
    const float* bn_v   = (const float*)d_in[8];
    const float* wq_w   = (const float*)d_in[9];
    const float* wq_b   = (const float*)d_in[10];
    const float* wk_w   = (const float*)d_in[11];
    const float* wk_b   = (const float*)d_in[12];
    const float* wv_w   = (const float*)d_in[13];
    const float* wv_b   = (const float*)d_in[14];
    const float* fc1_w  = (const float*)d_in[15];
    const float* fc1_b  = (const float*)d_in[16];
    const float* fc2_w  = (const float*)d_in[17];
    const float* fc2_b  = (const float*)d_in[18];
    const float* proj_w = (const float*)d_in[19];
    const float* proj_b = (const float*)d_in[20];
    float* out = (float*)d_out;

    char* ws = (char*)d_ws;
    size_t off = 0;
    auto carve = [&](size_t bytes) -> char* {
        char* p = ws + off;
        off += (bytes + 255) & ~(size_t)255;
        return p;
    };
    _Float16* Ff   = (_Float16*)carve((size_t)NS * NB * NPIX * CO * 2);  // 9.6 MB
    float*    af2  = (float*)   carve((size_t)NS * NB * NPIX * CO * 4);  // 19.3 MB
    _Float16* Qp   = (_Float16*)carve((size_t)NS * NB * NPIX * 32 * 2);  // 2.4 MB
    _Float16* Kt   = (_Float16*)carve((size_t)NS * NB * NPIX * 32 * 2);  // 2.4 MB
    _Float16* Vt   = (_Float16*)carve((size_t)NS * NB * NPIX * CO * 2);  // 9.6 MB
    float*    part = (float*)   carve((size_t)NS * NB * 8 * CO * 4);
    float*    cab  = (float*)   carve((size_t)NS * NB * CO * 4);
    _Float16* Wpw  = (_Float16*)carve((size_t)NS * CO * CI * 2);
    float*    bnA  = (float*)   carve((size_t)NS * CO * 4);
    float*    bnBv = (float*)   carve((size_t)NS * CO * 4);
    _Float16* Wqkv = (_Float16*)carve((size_t)NS * NQKV * CO * 2);
    float*    Bqkv = (float*)   carve((size_t)NS * NQKV * 4);
    _Float16* Wprj = (_Float16*)carve((size_t)CO * CI * 2);

    pack_kernel<<<dim3(240), 256, 0, stream>>>(
        pw_w, bn_g, bn_b, bn_m, bn_v,
        wq_w, wq_b, wk_w, wk_b, wv_w, wv_b, proj_w,
        Wpw, bnA, bnBv, Wqkv, Bqkv, Wprj);
    fixb_kernel<<<dim3((NS * CO + 127) / 128), 128, 0, stream>>>(pw_b, bnA, bnBv);
    feat_kernel<<<dim3(25, NB, NS), 256, 0, stream>>>(
        x, dw_w, dw_b, Wpw, bnA, bnBv, Ff);
    avgp_kernel<<<dim3(8, NS * NB), 128, 0, stream>>>(Ff, part);
    ca_kernel<<<dim3(NS * NB), 128, 0, stream>>>(part, fc1_w, fc1_b, fc2_w, fc2_b, cab);
    qkv_kernel<<<dim3(49, NB, NS), 256, 0, stream>>>(Ff, Wqkv, Bqkv, Qp, Kt, Vt);
    attn_kernel<<<dim3(25, NB, NS), 256, 0, stream>>>(Qp, Kt, Vt, cab, af2);
    fuse_kernel<<<dim3(25, NB), 256, 0, stream>>>(x, Wprj, proj_b, af2, out);
}
